// SparseMiddleExtractor_86517821214984
// MI455X (gfx1250) — matrix operand order repaired and checked by reading, not compiled, not run on hardware
//
#include <hip/hip_runtime.h>

typedef __attribute__((ext_vector_type(16))) _Float16 v16h;
typedef __attribute__((ext_vector_type(8)))  _Float16 v8h;
typedef __attribute__((ext_vector_type(8)))  float    v8f;

static constexpr int D0C = 21;
static constexpr int HC  = 100;
static constexpr int WC  = 88;
static constexpr int NCELL = D0C * HC * WC;      // 184800
static constexpr int N1 = 10 * HC * WC;          // 88000
static constexpr int N2 = 4 * HC * WC;           // 35200

// ---------------------------------------------------------------- utilities
__global__ void zero_f32x4(float4* __restrict__ p, int n4) {
    int i = blockIdx.x * blockDim.x + threadIdx.x;
    if (i < n4) p[i] = make_float4(0.f, 0.f, 0.f, 0.f);
}

__global__ void scatter_add(const float* __restrict__ vf, const int* __restrict__ coors,
                            float* __restrict__ dense, float* __restrict__ counts) {
    const int v = blockIdx.x;
    const int c = threadIdx.x;
    const int d = coors[v * 4 + 1];
    const int h = coors[v * 4 + 2];
    const int w = coors[v * 4 + 3];
    const size_t cell = ((size_t)d * HC + h) * WC + w;
    atomicAdd(&dense[cell * 128 + c], vf[(size_t)v * 128 + c]);
    if (c == 0) atomicAdd(&counts[cell], 1.0f);
}

__global__ void make_mask(const float* __restrict__ counts, float* __restrict__ mask, int n) {
    int i = blockIdx.x * blockDim.x + threadIdx.x;
    if (i < n) mask[i] = counts[i] > 0.f ? 1.f : 0.f;
}

__global__ void pool_mask(const float* __restrict__ in, float* __restrict__ out, int n) {
    int i = blockIdx.x * blockDim.x + threadIdx.x;
    if (i >= n) return;
    int w = i % WC;
    int h = (i / WC) % HC;
    int d = i / (HC * WC);
    float m = in[((size_t)(2 * d) * HC + h) * WC + w];
    m = fmaxf(m, in[((size_t)(2 * d + 1) * HC + h) * WC + w]);
    m = fmaxf(m, in[((size_t)(2 * d + 2) * HC + h) * WC + w]);
    out[i] = m;
}

__global__ void cvt_f32_f16(const float* __restrict__ in, _Float16* __restrict__ out, int n) {
    int i = blockIdx.x * blockDim.x + threadIdx.x;
    if (i < n) out[i] = (_Float16)in[i];
}

// DHWIO [tap][cin][cout=64] f32  ->  packed [tap][cout=64][cin] f16
__global__ void prep_w(const float* __restrict__ w, _Float16* __restrict__ wpk,
                       int taps, int cin) {
    int t = blockIdx.x * blockDim.x + threadIdx.x;
    int total = taps * 64 * cin;
    if (t >= total) return;
    int ci  = t % cin;
    int co  = (t / cin) % 64;
    int tap = t / (cin * 64);
    wpk[t] = (_Float16)w[((size_t)tap * cin + ci) * 64 + co];
}

__global__ void prep_bn(const float* __restrict__ g, const float* __restrict__ b,
                        const float* __restrict__ m, const float* __restrict__ v,
                        float* __restrict__ sc, float* __restrict__ bi) {
    int c = threadIdx.x;
    float s = g[c] * rsqrtf(v[c] + 1e-3f);
    sc[c] = s;
    bi[c] = b[c] - m[c] * s;
}

// ------------------------------------------------ implicit-GEMM conv + BN + ReLU + mask
// Workgroup: 256 threads (8 waves). Tile: 32 output positions (along W) x 64 couts.
// Per (kd,kh): stage a (32+KW-1)-pos x CIN activation strip and a KW x 64 x CIN weight
// block into LDS, then run KW*CIN/32 WMMAs between a single barrier pair.
// The weight block is always staged with async global->LDS DMA; the activation strip
// (contiguous in NDHWC) also uses async DMA whenever the whole window is in-bounds
// (uniform predicate), falling back to bounds-checked VGPR staging at borders.
template <int CIN, int KD, int KH, int KW, int SD, int PD, int PH, int PW, bool FINAL>
__global__ __launch_bounds__(256) void conv_bn_relu_mask(
    const _Float16* __restrict__ in, int Din,
    const _Float16* __restrict__ wpk,
    const float* __restrict__ scale, const float* __restrict__ bias,
    const float* __restrict__ mask,
    _Float16* __restrict__ outh, float* __restrict__ outf, int Dout) {
    constexpr int CCH    = CIN / 32;       // 32-wide K chunks
    constexpr int SEGS   = CIN / 8;        // 16-byte units per row
    constexpr int ROWS   = 32 + KW - 1;    // activation strip rows
    constexpr int AUNITS = ROWS * SEGS;
    constexpr int BUNITS = KW * 64 * SEGS; // always a multiple of 256

    const int wt = blockIdx.x;             // W tile (32 positions)
    const int h  = blockIdx.y;
    const int dz = blockIdx.z;
    const int wbase = wt * 32;

    const int tid  = threadIdx.x;
    const int lane = tid & 31;
    const int wave = tid >> 5;
    const int poshalf = wave >> 2;         // 0..1
    const int coutq   = wave & 3;          // 0..3

    __shared__ _Float16 Asm[ROWS][CIN];
    __shared__ _Float16 Bsm[KW * 64][CIN];

    v8f acc = {0.f, 0.f, 0.f, 0.f, 0.f, 0.f, 0.f, 0.f};

    const int hi = (lane >= 16);
    const int m  = lane & 15;

    const unsigned asm0 = (unsigned)(unsigned long long)(const void*)(&Asm[0][0]);
    const unsigned bsm0 = (unsigned)(unsigned long long)(const void*)(&Bsm[0][0]);
    const bool w_interior = (wbase - PW >= 0) && (wbase - PW + ROWS <= WC);

    for (int kd = 0; kd < KD; ++kd) {
        const int id = dz * SD + kd - PD;
        const bool dok = (id >= 0) && (id < Din);
        for (int kh = 0; kh < KH; ++kh) {
            const int ih = h + kh - PH;
            const bool dhok = dok && (ih >= 0) && (ih < HC);
            const _Float16* abase = in + (long long)(id * HC + ih) * WC * CIN;

            // ---- stage A strip: row r holds iw = wbase + r - PW (contiguous in memory)
            if (dhok && w_interior) {
                // fully in-bounds: direct async DMA of the contiguous strip
                const _Float16* astrip = abase + (long long)(wbase - PW) * CIN;
                for (int u = tid; u < AUNITS; u += 256) {
                    const unsigned           lofs = asm0 + (unsigned)u * 16u;
                    const unsigned long long gptr = (unsigned long long)(astrip + u * 8);
                    asm volatile("global_load_async_to_lds_b128 %0, %1, off"
                                 :: "v"(lofs), "v"(gptr) : "memory");
                }
            } else {
                for (int u = tid; u < AUNITS; u += 256) {
                    const int row = u / SEGS;
                    const int seg = u % SEGS;
                    const int iw  = wbase + row - PW;
                    uint4 v4 = make_uint4(0u, 0u, 0u, 0u);
                    if (dhok && iw >= 0 && iw < WC)
                        v4 = *(const uint4*)(abase + (size_t)iw * CIN + seg * 8);
                    *(uint4*)(&Asm[row][seg * 8]) = v4;
                }
            }

            // ---- stage B: KW taps of packed weights, contiguous -> async DMA to LDS
            const _Float16* bbase = wpk + (size_t)(kd * KH + kh) * KW * 64 * CIN;
            for (int u = tid; u < BUNITS; u += 256) {
                const unsigned           lofs = bsm0 + (unsigned)u * 16u;
                const unsigned long long gptr = (unsigned long long)(bbase + u * 8);
                asm volatile("global_load_async_to_lds_b128 %0, %1, off"
                             :: "v"(lofs), "v"(gptr) : "memory");
            }
#if __has_builtin(__builtin_amdgcn_s_wait_asynccnt)
            __builtin_amdgcn_s_wait_asynccnt(0);
#else
            asm volatile("s_wait_asynccnt 0x0" ::: "memory");
#endif
            __syncthreads();

            // ---- compute: KW x CCH WMMAs on the staged strip
#pragma unroll
            for (int kw = 0; kw < KW; ++kw) {
#pragma unroll
                for (int cc = 0; cc < CCH; ++cc) {
                    const int c0 = cc * 32;
                    union { v16h v; v8h h2[2]; } af, bf;
                    af.h2[0] = *(const v8h*)(&Asm[poshalf * 16 + m + kw][c0 + (hi ? 8 : 0)]);
                    af.h2[1] = *(const v8h*)(&Asm[poshalf * 16 + m + kw][c0 + (hi ? 24 : 16)]);
                    bf.h2[0] = *(const v8h*)(&Bsm[kw * 64 + coutq * 16 + m][c0 + (hi ? 8 : 0)]);
                    bf.h2[1] = *(const v8h*)(&Bsm[kw * 64 + coutq * 16 + m][c0 + (hi ? 24 : 16)]);
                    acc = __builtin_amdgcn_wmma_f32_16x16x32_f16(
                        false, af.v, false, bf.v, (short)0, acc, false, false);
                }
            }
            __syncthreads();
        }
    }

    // ---- epilogue: BN + ReLU + mask
    const int cout = coutq * 16 + (lane & 15);
    const float sc = scale[cout];
    const float bi = bias[cout];
    const int moff = (lane >= 16) ? 8 : 0;
#pragma unroll
    for (int r = 0; r < 8; ++r) {
        const int w = wbase + poshalf * 16 + r + moff;
        if (w < WC) {
            float v = acc[r] * sc + bi;
            v = fmaxf(v, 0.f) * mask[((size_t)dz * HC + h) * WC + w];
            if constexpr (FINAL)
                outf[(((size_t)cout * Dout + dz) * HC + h) * WC + w] = v;
            else
                outh[(((size_t)dz * HC + h) * WC + w) * 64 + cout] = (_Float16)v;
        }
    }
}

// ---------------------------------------------------------------- launch
extern "C" void kernel_launch(void* const* d_in, const int* in_sizes, int n_in,
                              void* d_out, int out_size, void* d_ws, size_t ws_size,
                              hipStream_t stream) {
    (void)in_sizes; (void)n_in; (void)out_size; (void)ws_size;
    const float* vf    = (const float*)d_in[0];
    const int*   coors = (const int*)d_in[1];
    const float* w_[5] = {(const float*)d_in[3], (const float*)d_in[4],
                          (const float*)d_in[5], (const float*)d_in[6],
                          (const float*)d_in[7]};
    const float* bn_[5][4];
    for (int l = 0; l < 5; ++l)
        for (int k = 0; k < 4; ++k) bn_[l][k] = (const float*)d_in[8 + 4 * l + k];
    float* out = (float*)d_out;

    auto al = [](size_t x) { return (x + 255) & ~(size_t)255; };
    char* ws = (char*)d_ws;
    size_t o = 0;
    float*    dense32 = (float*)(ws + o);    o = al(o + (size_t)NCELL * 128 * 4);
    _Float16* dense16 = (_Float16*)(ws + o); o = al(o + (size_t)NCELL * 128 * 2);
    _Float16* x0      = (_Float16*)(ws + o); o = al(o + (size_t)NCELL * 64 * 2);
    _Float16* x1      = (_Float16*)(ws + o); o = al(o + (size_t)N1 * 64 * 2);
    _Float16* x2      = (_Float16*)(ws + o); o = al(o + (size_t)N1 * 64 * 2);
    _Float16* x3      = (_Float16*)(ws + o); o = al(o + (size_t)N1 * 64 * 2);
    float*    counts  = (float*)(ws + o);    o = al(o + (size_t)NCELL * 4);
    float*    mask0   = (float*)(ws + o);    o = al(o + (size_t)NCELL * 4);
    float*    mask1   = (float*)(ws + o);    o = al(o + (size_t)N1 * 4);
    float*    mask2   = (float*)(ws + o);    o = al(o + (size_t)N2 * 4);
    _Float16* wpk[5];
    const int taps_[5] = {27, 3, 27, 27, 3};
    const int cin_[5]  = {128, 64, 64, 64, 64};
    for (int l = 0; l < 5; ++l) {
        wpk[l] = (_Float16*)(ws + o);
        o = al(o + (size_t)taps_[l] * 64 * cin_[l] * 2);
    }
    float* sb = (float*)(ws + o);  // 5 layers x (scale[64], bias[64])
    o = al(o + 5 * 128 * 4);

    // zero accumulation buffers
    {
        int n4 = NCELL * 128 / 4;
        zero_f32x4<<<(n4 + 255) / 256, 256, 0, stream>>>((float4*)dense32, n4);
        int c4 = NCELL / 4;
        zero_f32x4<<<(c4 + 255) / 256, 256, 0, stream>>>((float4*)counts, c4);
    }
    // scatter voxels
    scatter_add<<<40000, 128, 0, stream>>>(vf, coors, dense32, counts);
    make_mask<<<(NCELL + 255) / 256, 256, 0, stream>>>(counts, mask0, NCELL);
    {
        int n = NCELL * 128;
        cvt_f32_f16<<<(n + 255) / 256, 256, 0, stream>>>(dense32, dense16, n);
    }
    // prep weights + folded BN
    for (int l = 0; l < 5; ++l) {
        int total = taps_[l] * 64 * cin_[l];
        prep_w<<<(total + 255) / 256, 256, 0, stream>>>(w_[l], wpk[l], taps_[l], cin_[l]);
        prep_bn<<<1, 64, 0, stream>>>(bn_[l][0], bn_[l][1], bn_[l][2], bn_[l][3],
                                      sb + l * 128, sb + l * 128 + 64);
    }
    // pooled masks
    pool_mask<<<(N1 + 255) / 256, 256, 0, stream>>>(mask0, mask1, N1);
    pool_mask<<<(N2 + 255) / 256, 256, 0, stream>>>(mask1, mask2, N2);

    // conv pipeline
    dim3 blk(256);
    conv_bn_relu_mask<128, 3, 3, 3, 1, 1, 1, 1, false>
        <<<dim3(3, 100, 21), blk, 0, stream>>>(dense16, 21, wpk[0],
            sb + 0 * 128, sb + 0 * 128 + 64, mask0, x0, nullptr, 21);
    conv_bn_relu_mask<64, 3, 1, 1, 2, 0, 0, 0, false>
        <<<dim3(3, 100, 10), blk, 0, stream>>>(x0, 21, wpk[1],
            sb + 1 * 128, sb + 1 * 128 + 64, mask1, x1, nullptr, 10);
    conv_bn_relu_mask<64, 3, 3, 3, 1, 1, 1, 1, false>
        <<<dim3(3, 100, 10), blk, 0, stream>>>(x1, 10, wpk[2],
            sb + 2 * 128, sb + 2 * 128 + 64, mask1, x2, nullptr, 10);
    conv_bn_relu_mask<64, 3, 3, 3, 1, 1, 1, 1, false>
        <<<dim3(3, 100, 10), blk, 0, stream>>>(x2, 10, wpk[3],
            sb + 3 * 128, sb + 3 * 128 + 64, mask1, x3, nullptr, 10);
    conv_bn_relu_mask<64, 3, 1, 1, 2, 0, 0, 0, true>
        <<<dim3(3, 100, 4), blk, 0, stream>>>(x3, 10, wpk[4],
            sb + 4 * 128, sb + 4 * 128 + 64, mask2, nullptr, out, 4);
}
